// DocLLMAttention_41936060678484
// MI455X (gfx1250) — compile-verified, matrix operand order checked by reading, not run on hardware
//
#include <hip/hip_runtime.h>

// DocLLM attention for MI455X (gfx1250): f16 WMMA 16x16x32 everywhere,
// TDM (tensor_load_to_lds) double-buffered GEMM staging, flash attention.

#define B_    2
#define S_    2048
#define H_    2048
#define NH_   32
#define NKV_  8
#define HD_   64
#define GROUPS_ 4
#define SCALE_ 0.125f   // 64^-0.5

typedef __attribute__((ext_vector_type(16))) _Float16 v16h;
typedef __attribute__((ext_vector_type(8)))  _Float16 v8h;
typedef __attribute__((ext_vector_type(8)))  float    v8f;
typedef __attribute__((ext_vector_type(4)))  unsigned int u32x4;
typedef __attribute__((ext_vector_type(4)))  int      i32x4;
typedef __attribute__((ext_vector_type(8)))  int      i32x8;

#if __has_builtin(__builtin_amdgcn_tensor_load_to_lds) && \
    __has_builtin(__builtin_amdgcn_s_wait_tensorcnt)
#define HAVE_TDM 1
#else
#define HAVE_TDM 0
#endif

// ---------------------------------------------------------------------------
// Fragment loader matching the CDNA5 16-bit A/B WMMA VGPR layout:
//   lanes 0-15 : K = {0..7, 16..23}; lanes 16-31: K = {8..15, 24..31}
// Source rows must have the fragment's K-values contiguous (row-major A /
// N-major B^T / LDS tiles with ld=32). Two 16-byte loads per lane.
// ---------------------------------------------------------------------------
__device__ __forceinline__ v16h frag_load(const _Float16* base,
                                          int row, int ld, int kb, int half) {
  const _Float16* p = base + (size_t)row * (size_t)ld + kb + half * 8;
  v8h lo = *(const v8h*)p;
  v8h hi = *(const v8h*)(p + 16);
  return __builtin_shufflevector(lo, hi, 0,1,2,3,4,5,6,7,8,9,10,11,12,13,14,15);
}

__device__ __forceinline__ v8f wmma16(v16h a, v16h b, v8f c) {
  return __builtin_amdgcn_wmma_f32_16x16x32_f16(false, a, false, b,
                                                (short)0, c, false, false);
}

#if HAVE_TDM
// ---------------------------------------------------------------------------
// Issue one TDM 2D tile load: rows x 32 f16 elements, row stride = strideElems.
// D# per CDNA5 ISA ch.7/8: group0 = {count|flags, lds_addr, global_addr, type},
// group1 = {mask/data_size, dims/tile dims, tensor_dim0_stride}.
// ---------------------------------------------------------------------------
__device__ __forceinline__ void tdm_load_tile(unsigned int lds_off,
                                              const _Float16* gptr,
                                              unsigned int rows,
                                              unsigned long long strideElems) {
  unsigned long long ga = (unsigned long long)(size_t)gptr;
  u32x4 g0;
  g0[0] = 1u;                                         // count=1, user mode
  g0[1] = lds_off;                                    // LDS byte address
  g0[2] = (unsigned int)(ga & 0xffffffffu);           // global_addr[31:0]
  g0[3] = (unsigned int)((ga >> 32) & 0x01ffffffu) | (2u << 30); // [56:32]|type=2
  unsigned long long st = strideElems;
  i32x8 g1;
  g1[0] = (int)(1u << 16);                            // data_size=1 -> 2 bytes
  g1[1] = (int)((32u & 0xffffu) << 16);               // tensor_dim0 = 32 (lo16)
  g1[2] = (int)((rows & 0xffffu) << 16);              // dim0 hi=0 | tensor_dim1 lo
  g1[3] = (int)(32u << 16);                           // tensor_dim1 hi=0 | tile_dim0=32
  g1[4] = (int)(rows & 0xffffu);                      // tile_dim1 | tile_dim2=0
  g1[5] = (int)(unsigned int)(st & 0xffffffffu);      // tensor_dim0_stride[31:0]
  g1[6] = (int)(unsigned int)((st >> 32) & 0xffffu);  // stride[47:32] | stride1 lo=0
  g1[7] = 0;
  i32x4 z4 = {0, 0, 0, 0};
#if defined(__clang_major__) && (__clang_major__ >= 23)
  i32x8 z8 = {0, 0, 0, 0, 0, 0, 0, 0};
  __builtin_amdgcn_tensor_load_to_lds(g0, g1, z4, z4, z8, 0);
#else
  __builtin_amdgcn_tensor_load_to_lds(g0, g1, z4, z4, 0);
#endif
}
#endif

// ---------------------------------------------------------------------------
// Elementwise f32 -> f16
// ---------------------------------------------------------------------------
__global__ void cvt_f16(const float* __restrict__ x, _Float16* __restrict__ y,
                        size_t n) {
  size_t i = (size_t)blockIdx.x * blockDim.x + threadIdx.x;
  if (i < n) y[i] = (_Float16)x[i];
}

// W [K][N] f32 (row-major) -> WT [N][K] f16 (N-major, WMMA B operand)
__global__ void wt_f16(const float* __restrict__ W, _Float16* __restrict__ WT,
                       int K, int N) {
  size_t i = (size_t)blockIdx.x * blockDim.x + threadIdx.x;
  if (i >= (size_t)K * N) return;
  int k = (int)(i / N), n = (int)(i % N);
  WT[(size_t)n * K + k] = (_Float16)W[i];
}

// ---------------------------------------------------------------------------
// Tiled WMMA GEMM: C[M][N] = A[M][K] @ B[K][N], A row-major f16,
// Bt = B^T in [N][K] f16. Block = 256 threads = 8 waves; block tile 128x64.
// A (128x32) and B (64x32) K-slices are staged in LDS by the Tensor Data
// Mover, double-buffered: wave 0 issues TDM for slice it+1, waits TENSORcnt
// for slice it, then a block barrier publishes the buffer. Compute waves do
// only ds_load_b128 + v_wmma per iteration.
// ---------------------------------------------------------------------------
__global__ __launch_bounds__(256)
void gemm_wmma(const _Float16* __restrict__ A, const _Float16* __restrict__ Bt,
               void* __restrict__ C, int M, int N, int K, int c_is_f32) {
  __shared__ _Float16 aLds[2][128 * 32];   // 8KB x2
  __shared__ _Float16 bLds[2][64 * 32];    // 4KB x2
  const int wid  = threadIdx.x >> 5;
  const int lane = threadIdx.x & 31;
  const int half = lane >> 4;
  const int r    = lane & 15;
  const int row0 = blockIdx.y * 128;
  const int col0 = blockIdx.x * 64;
  const int nIter = K >> 5;

#if HAVE_TDM
  if (wid == 0) {
    tdm_load_tile((unsigned int)(size_t)&aLds[0][0], A + (size_t)row0 * K, 128,
                  (unsigned long long)K);
    tdm_load_tile((unsigned int)(size_t)&bLds[0][0], Bt + (size_t)col0 * K, 64,
                  (unsigned long long)K);
  }
#endif

  v8f acc[4] = {};
  for (int it = 0; it < nIter; ++it) {
    const int cur = it & 1;
#if HAVE_TDM
    if (wid == 0) {
      if (it + 1 < nIter) {
        // prefetch next slice into the other buffer, then wait only for the
        // current slice's two TDM ops (in-order completion per wave)
        tdm_load_tile((unsigned int)(size_t)&aLds[cur ^ 1][0],
                      A + (size_t)row0 * K + (it + 1) * 32, 128,
                      (unsigned long long)K);
        tdm_load_tile((unsigned int)(size_t)&bLds[cur ^ 1][0],
                      Bt + (size_t)col0 * K + (it + 1) * 32, 64,
                      (unsigned long long)K);
        __builtin_amdgcn_s_wait_tensorcnt(2);
      } else {
        __builtin_amdgcn_s_wait_tensorcnt(0);
      }
    }
#else
    // Fallback: cooperative vector copy (256 threads)
    {
      const _Float16* ga = A + (size_t)row0 * K + it * 32;
      const _Float16* gb = Bt + (size_t)col0 * K + it * 32;
      int tid = threadIdx.x;
      // A tile: 128 rows x 32 -> 512 x 8-elem chunks; 2 per thread
      for (int c = 0; c < 2; ++c) {
        int chunk = tid + c * 256;
        int rr = chunk >> 2, cc = (chunk & 3) * 8;
        *(v8h*)&aLds[cur][rr * 32 + cc] = *(const v8h*)(ga + (size_t)rr * K + cc);
      }
      // B tile: 64 rows x 32 -> 256 chunks; 1 per thread
      {
        int rr = tid >> 2, cc = (tid & 3) * 8;
        *(v8h*)&bLds[cur][rr * 32 + cc] = *(const v8h*)(gb + (size_t)rr * K + cc);
      }
    }
#endif
    __syncthreads();
    v16h a = frag_load(&aLds[cur][0], wid * 16 + r, 32, 0, half);
#pragma unroll
    for (int t = 0; t < 4; ++t) {
      v16h b = frag_load(&bLds[cur][0], t * 16 + r, 32, 0, half);
      acc[t] = wmma16(a, b, acc[t]);
    }
    __syncthreads();   // all waves done reading before buffer is re-staged
  }

  // C fragment layout: VGPR v -> M = v + 8*half, N = r
#pragma unroll
  for (int t = 0; t < 4; ++t) {
#pragma unroll
    for (int v = 0; v < 8; ++v) {
      size_t mm = (size_t)(row0 + wid * 16 + v + 8 * half);
      size_t nn = (size_t)(col0 + t * 16 + r);
      if (c_is_f32) ((float*)C)[mm * N + nn] = acc[t][v];
      else          ((_Float16*)C)[mm * N + nn] = (_Float16)acc[t][v];
    }
  }
}

// ---------------------------------------------------------------------------
// RoPE is linear, so rope(q)+rope(sq) = rope(q+sq). Fuse add + rope + reshape
// [B,S,nh*64] -> [B,nh,S,64].
// ---------------------------------------------------------------------------
__global__ void rope_add(const _Float16* __restrict__ xp,
                         const _Float16* __restrict__ sp,
                         const float* __restrict__ cosb,
                         const float* __restrict__ sinb,
                         _Float16* __restrict__ out, int nh) {
  size_t i = (size_t)blockIdx.x * blockDim.x + threadIdx.x;
  size_t total = (size_t)B_ * S_ * nh * HD_;
  if (i >= total) return;
  int d = (int)(i & 63);
  int h = (int)((i >> 6) % nh);
  int s = (int)((i / ((size_t)nh * HD_)) % S_);
  int b = (int)(i / ((size_t)nh * HD_ * S_));
  size_t rowi = (size_t)b * S_ + s;
  size_t src  = rowi * ((size_t)nh * HD_) + (size_t)h * HD_;
  int   d2  = (d < 32) ? d + 32 : d - 32;
  float sgn = (d < 32) ? -1.f : 1.f;
  float c  = cosb[rowi * HD_ + d];
  float sn = sinb[rowi * HD_ + d];
  float x  = (float)xp[src + d]  + (float)sp[src + d];
  float x2 = (float)xp[src + d2] + (float)sp[src + d2];
  out[(((size_t)b * nh + h) * S_ + s) * HD_ + d] = (_Float16)(x * c + sgn * x2 * sn);
}

// vp [B,S,NKV*64] -> vt [B,NKV,64,S]  (V^T so PV's B-fragment reads are contiguous)
__global__ void v_transpose(const _Float16* __restrict__ vp,
                            _Float16* __restrict__ vt) {
  size_t i = (size_t)blockIdx.x * blockDim.x + threadIdx.x;
  size_t total = (size_t)B_ * S_ * NKV_ * HD_;
  if (i >= total) return;
  int d  = (int)(i & 63);
  int kv = (int)((i >> 6) % NKV_);
  int s  = (int)((i / (NKV_ * HD_)) % S_);
  int b  = (int)(i / ((size_t)NKV_ * HD_ * S_));
  vt[(((size_t)b * NKV_ + kv) * HD_ + d) * S_ + s] =
      vp[((size_t)b * S_ + s) * (NKV_ * HD_) + (size_t)kv * HD_ + d];
}

// ---------------------------------------------------------------------------
// Flash attention, causal, GQA. One wave per 16-query tile of one (b,h).
// Causal trip counts are wave-divergent, so no block barriers: direct global
// fragment loads, wave-private LDS round-trip for the P reshuffle.
// ---------------------------------------------------------------------------
__global__ __launch_bounds__(128)
void attn_flash(const _Float16* __restrict__ qr,   // [B][NH][S][64]
                const _Float16* __restrict__ kr,   // [B][NKV][S][64]
                const _Float16* __restrict__ vt,   // [B][NKV][64][S]
                _Float16* __restrict__ attn) {     // [B][S][NH*64]
  __shared__ _Float16 plds[4 * 16 * 32];
  const int wid  = threadIdx.x >> 5;
  const int lane = threadIdx.x & 31;
  const int half = lane >> 4;
  const int r    = lane & 15;
  const int b  = blockIdx.z;
  const int h  = blockIdx.y;
  const int kv = h / GROUPS_;
  const int q0 = (blockIdx.x * 4 + wid) * 16;

  const _Float16* qbase = qr + (((size_t)b * NH_  + h ) * S_) * HD_;
  const _Float16* kbase = kr + (((size_t)b * NKV_ + kv) * S_) * HD_;
  const _Float16* vbase = vt + (((size_t)b * NKV_ + kv) * HD_) * S_;
  _Float16* pl = plds + wid * 16 * 32;

  v16h qa0 = frag_load(qbase, q0 + r, HD_, 0,  half);
  v16h qa1 = frag_load(qbase, q0 + r, HD_, 32, half);

  v8f acc[4] = {};
  float mrow[8], lrow[8];
#pragma unroll
  for (int v = 0; v < 8; ++v) { mrow[v] = -1e30f; lrow[v] = 0.f; }

  const int kend = q0 + 16;                 // causal bound (exclusive)
  for (int kb = 0; kb < kend; kb += 32) {
    // ---- scores: 16x32 block of QK^T (two 16x16 WMMA tiles) ----
    v16h k00 = frag_load(kbase, kb + r,      HD_, 0,  half);
    v16h k01 = frag_load(kbase, kb + r,      HD_, 32, half);
    v16h k10 = frag_load(kbase, kb + 16 + r, HD_, 0,  half);
    v16h k11 = frag_load(kbase, kb + 16 + r, HD_, 32, half);
    v8f s0 = {}, s1 = {};
    s0 = wmma16(qa0, k00, s0); s0 = wmma16(qa1, k01, s0);
    s1 = wmma16(qa0, k10, s1); s1 = wmma16(qa1, k11, s1);

    // ---- online softmax (row-wise over 16-lane halves) ----
    float p0[8], p1[8], alpha[8];
#pragma unroll
    for (int v = 0; v < 8; ++v) {
      int qi  = q0 + v + 8 * half;
      int kj0 = kb + r, kj1 = kb + 16 + r;
      float x0 = (kj0 <= qi) ? s0[v] * SCALE_ : -1e30f;
      float x1 = (kj1 <= qi) ? s1[v] * SCALE_ : -1e30f;
      float rm = fmaxf(x0, x1);
#pragma unroll
      for (int off = 1; off < 16; off <<= 1)
        rm = fmaxf(rm, __shfl_xor(rm, off, 32));
      float mnew = fmaxf(mrow[v], rm);
      alpha[v] = __expf(mrow[v] - mnew);
      mrow[v]  = mnew;
      float e0 = __expf(x0 - mnew);
      float e1 = __expf(x1 - mnew);
      float rs = e0 + e1;
#pragma unroll
      for (int off = 1; off < 16; off <<= 1)
        rs += __shfl_xor(rs, off, 32);
      lrow[v] = lrow[v] * alpha[v] + rs;
      p0[v] = e0; p1[v] = e1;
    }
#pragma unroll
    for (int t = 0; t < 4; ++t)
#pragma unroll
      for (int v = 0; v < 8; ++v) acc[t][v] *= alpha[v];

    // ---- C-layout P -> LDS -> A-fragment (cross-lane reshuffle) ----
#pragma unroll
    for (int v = 0; v < 8; ++v) {
      int row = v + 8 * half;
      pl[row * 32 + r]      = (_Float16)p0[v];
      pl[row * 32 + 16 + r] = (_Float16)p1[v];
    }
    __threadfence_block();               // order DS store -> DS load (in-wave)
    v16h pf = frag_load(pl, r, 32, 0, half);

    // ---- acc += P @ V  (V^T rows are d, contiguous along keys) ----
#pragma unroll
    for (int t = 0; t < 4; ++t) {
      v16h vf = frag_load(vbase, t * 16 + r, S_, kb, half);
      acc[t] = wmma16(pf, vf, acc[t]);
    }
    __threadfence_block();               // WAR: next iter rewrites pl
  }

  // ---- normalize + scatter to [B,S,NH*64] for the Wo GEMM ----
#pragma unroll
  for (int t = 0; t < 4; ++t) {
#pragma unroll
    for (int v = 0; v < 8; ++v) {
      int qi = q0 + v + 8 * half;
      int d  = t * 16 + r;
      attn[((size_t)b * S_ + qi) * (NH_ * HD_) + (size_t)h * HD_ + d] =
          (_Float16)(acc[t][v] / lrow[v]);
    }
  }
}

// ---------------------------------------------------------------------------
// Host-side launch
// ---------------------------------------------------------------------------
extern "C" void kernel_launch(void* const* d_in, const int* in_sizes, int n_in,
                              void* d_out, int out_size, void* d_ws, size_t ws_size,
                              hipStream_t stream) {
  (void)in_sizes; (void)n_in; (void)out_size; (void)ws_size;
  const float* hs   = (const float*)d_in[0];
  const float* bbox = (const float*)d_in[1];
  const float* cosb = (const float*)d_in[2];
  const float* sinb = (const float*)d_in[3];
  const float* Wq   = (const float*)d_in[4];
  const float* Wk   = (const float*)d_in[5];
  const float* Wv   = (const float*)d_in[6];
  const float* Wo   = (const float*)d_in[7];
  const float* Wsq  = (const float*)d_in[8];
  const float* Wsk  = (const float*)d_in[9];

  const size_t BS   = (size_t)B_ * S_;          // 4096 rows
  const size_t NQD  = (size_t)NH_ * HD_;        // 2048
  const size_t NKD  = (size_t)NKV_ * HD_;       // 512
  const size_t eAct = BS * H_;                  // 8.39M
  const size_t eKV  = BS * NKD;                 // 2.10M

  char* wsp = (char*)d_ws;
  auto carve = [&](size_t elems) {
    _Float16* q = (_Float16*)wsp;
    wsp += (elems * sizeof(_Float16) + 255) & ~(size_t)255;
    return q;
  };
  _Float16* hsH  = carve(eAct);
  _Float16* bbH  = carve(eAct);
  _Float16* WqT  = carve((size_t)H_ * NQD);
  _Float16* WkT  = carve((size_t)H_ * NKD);
  _Float16* WvT  = carve((size_t)H_ * NKD);
  _Float16* WoT  = carve((size_t)NQD * H_);
  _Float16* WsqT = carve((size_t)H_ * NQD);
  _Float16* WskT = carve((size_t)H_ * NKD);
  _Float16* qp   = carve(BS * NQD);
  _Float16* kp   = carve(eKV);
  _Float16* vp   = carve(eKV);
  _Float16* sqp  = carve(BS * NQD);
  _Float16* skp  = carve(eKV);
  _Float16* qrr  = carve(BS * NQD);
  _Float16* krr  = carve(eKV);
  _Float16* vtt  = carve(eKV);
  _Float16* atn  = carve(BS * NQD);

  const int T = 256;
  auto nb = [&](size_t n) { return (unsigned)((n + T - 1) / T); };

  // 1) precision / layout conversion (single pass; everything lives in L2)
  cvt_f16<<<nb(eAct), T, 0, stream>>>(hs,   hsH, eAct);
  cvt_f16<<<nb(eAct), T, 0, stream>>>(bbox, bbH, eAct);
  wt_f16<<<nb((size_t)H_ * NQD), T, 0, stream>>>(Wq,  WqT,  H_,  (int)NQD);
  wt_f16<<<nb((size_t)H_ * NKD), T, 0, stream>>>(Wk,  WkT,  H_,  (int)NKD);
  wt_f16<<<nb((size_t)H_ * NKD), T, 0, stream>>>(Wv,  WvT,  H_,  (int)NKD);
  wt_f16<<<nb((size_t)NQD * H_), T, 0, stream>>>(Wo,  WoT,  (int)NQD, H_);
  wt_f16<<<nb((size_t)H_ * NQD), T, 0, stream>>>(Wsq, WsqT, H_,  (int)NQD);
  wt_f16<<<nb((size_t)H_ * NKD), T, 0, stream>>>(Wsk, WskT, H_,  (int)NKD);

  // 2) projections (TDM-staged WMMA GEMMs)
  dim3 gq((unsigned)(NQD / 64), (unsigned)(BS / 128));
  dim3 gk((unsigned)(NKD / 64), (unsigned)(BS / 128));
  gemm_wmma<<<gq, 256, 0, stream>>>(hsH, WqT,  qp,  (int)BS, (int)NQD, H_, 0);
  gemm_wmma<<<gk, 256, 0, stream>>>(hsH, WkT,  kp,  (int)BS, (int)NKD, H_, 0);
  gemm_wmma<<<gk, 256, 0, stream>>>(hsH, WvT,  vp,  (int)BS, (int)NKD, H_, 0);
  gemm_wmma<<<gq, 256, 0, stream>>>(bbH, WsqT, sqp, (int)BS, (int)NQD, H_, 0);
  gemm_wmma<<<gk, 256, 0, stream>>>(bbH, WskT, skp, (int)BS, (int)NKD, H_, 0);

  // 3) fused add + RoPE + head-major reshape; V transpose
  rope_add<<<nb(BS * NQD), T, 0, stream>>>(qp, sqp, cosb, sinb, qrr, NH_);
  rope_add<<<nb(eKV),      T, 0, stream>>>(kp, skp, cosb, sinb, krr, NKV_);
  v_transpose<<<nb(eKV),   T, 0, stream>>>(vp, vtt);

  // 4) causal flash attention (WMMA QK^T and PV)
  dim3 ga((unsigned)(S_ / 16 / 4), NH_, B_);
  attn_flash<<<ga, 128, 0, stream>>>(qrr, krr, vtt, atn);

  // 5) output projection, f32 result
  gemm_wmma<<<gq, 256, 0, stream>>>(atn, WoT, d_out, (int)BS, H_, (int)NQD, 1);
}
